// DivGraphNet_33621003993196
// MI455X (gfx1250) — compile-verified, matrix operand order checked by reading, not run on hardware
//
#include <hip/hip_runtime.h>
#include <hip/hip_bf16.h>
#include <math.h>

// ---------------------------------------------------------------------------
// DivGraphNet decoder for MI455X (gfx1250, wave32).
//
// Roofline:
//  * nodes@An : 68.7 GFLOP GEMM -> v_wmma_f32_16x16x32_bf16, wave tile 16x64
//    (4 accumulators / A-frag reuse); block tile 16x512 so A streams once:
//    HBM ~= 256MB(A) + 128MB(C-bf16) ~ 16us.
//  * 55 sequential decode steps stream nodes_proj bf16 (128 MB/step):
//    ~0.3 ms total @ 23.3 TB/s -> the true floor; attention kernel uses
//    u16x2 loads (128B/wave/instr) + stride-2 conflict-free LDS tables.
//  * all small GEMMs (K=512 throughout) routed through the same WMMA kernel.
// Output: d_out[0..3839] = kps (as float(wid)), d_out[3840] = loss.
// ---------------------------------------------------------------------------

#define Bn 64
#define Nn 2048
#define En 512
#define Hn 512
#define KP 5
#define Tn 11
#define Kc 512   // every GEMM in this model has K = 512

typedef unsigned short u16;
typedef __attribute__((ext_vector_type(8)))  float  f32x8;
typedef __attribute__((ext_vector_type(16))) __bf16 bf16x16;
typedef __attribute__((ext_vector_type(2)))  u16    u16x2;
typedef __attribute__((ext_vector_type(8)))  u16    u16x8;
typedef __attribute__((ext_vector_type(16))) u16    u16x16;

__device__ __forceinline__ u16 f32_to_bf16(float f) {
    unsigned int u = __float_as_uint(f);
    unsigned int r = (u + 0x7FFFu + ((u >> 16) & 1u)) >> 16;   // RNE
    return (u16)r;
}
__device__ __forceinline__ float bf16_to_f32(u16 h) {
    return __uint_as_float(((unsigned int)h) << 16);
}
__device__ __forceinline__ float sigmoidf_(float x) { return 1.0f / (1.0f + __expf(-x)); }

// A fragment: 16x32 bf16 from f32 row-major [M,Kc].
// ISA 16-bit A layout: lanes 0-15 -> row M=lane, K=0..7 & 16..23 ;
//                      lanes 16-31 -> same rows, K=8..15 & 24..31.
__device__ __forceinline__ bf16x16 load_a_frag_f32(const float* A, int row, int kb) {
    f32x8 lo = *(const f32x8*)(A + (size_t)row * Kc + kb);
    f32x8 hi = *(const f32x8*)(A + (size_t)row * Kc + kb + 16);
    u16x16 u;
#pragma unroll
    for (int i = 0; i < 8; ++i) { u[i] = f32_to_bf16(lo[i]); u[8 + i] = f32_to_bf16(hi[i]); }
    return __builtin_bit_cast(bf16x16, u);
}
// B fragment from pre-transposed bf16 weights Bt[Nd,Kc] (lane holds column N=lane&15).
__device__ __forceinline__ bf16x16 load_b_frag_bf16(const u16* Bt, int col, int kb) {
    u16x8 lo = *(const u16x8*)(Bt + (size_t)col * Kc + kb);
    u16x8 hi = *(const u16x8*)(Bt + (size_t)col * Kc + kb + 16);
    u16x16 u = __builtin_shufflevector(lo, hi, 0,1,2,3,4,5,6,7,8,9,10,11,12,13,14,15);
    return __builtin_bit_cast(bf16x16, u);
}

// ---------------- WMMA GEMM: C[M,Nd] f32 = A[M,512] f32 * Bt[Nd,512]^T (+bias, opt tanh)
// wave tile: 16(M) x 64(N), 4 accumulators sharing one A fragment.
// tile linearization n-band-major: the 8 waves of a block cover one 16x512 band
// of C, so the A fragment is block-uniform (single L0 stream of A).
__global__ __launch_bounds__(256) void
wmma_gemm_f32(const float* __restrict__ A, const u16* __restrict__ Bt,
              const float* __restrict__ bias, float* __restrict__ C,
              int M, int Nd, int act) {
    int wave = threadIdx.x >> 5, lane = threadIdx.x & 31;
    int bandsN = Nd >> 6;                         // 64-wide bands
    int tile = blockIdx.x * 8 + wave;
    if (tile >= (M >> 4) * bandsN) return;        // wave-uniform guard (EXEC stays full)
    int m0 = (tile / bandsN) << 4, n0 = (tile % bandsN) << 6;
    int row  = m0 + (lane & 15);
    int colb = n0 + (lane & 15);
    int koff = (lane >> 4) << 3;                  // 0 or 8 per ISA 16-bit A layout
    f32x8 c0 = {}, c1 = {}, c2 = {}, c3 = {};
#pragma unroll 2
    for (int k0 = 0; k0 < Kc; k0 += 32) {
        bf16x16 a  = load_a_frag_f32(A, row, k0 + koff);
        bf16x16 b0 = load_b_frag_bf16(Bt, colb,      k0 + koff);
        bf16x16 b1 = load_b_frag_bf16(Bt, colb + 16, k0 + koff);
        bf16x16 b2 = load_b_frag_bf16(Bt, colb + 32, k0 + koff);
        bf16x16 b3 = load_b_frag_bf16(Bt, colb + 48, k0 + koff);
        c0 = __builtin_amdgcn_wmma_f32_16x16x32_bf16(false, a, false, b0, (short)0, c0, false, false);
        c1 = __builtin_amdgcn_wmma_f32_16x16x32_bf16(false, a, false, b1, (short)0, c1, false, false);
        c2 = __builtin_amdgcn_wmma_f32_16x16x32_bf16(false, a, false, b2, (short)0, c2, false, false);
        c3 = __builtin_amdgcn_wmma_f32_16x16x32_bf16(false, a, false, b3, (short)0, c3, false, false);
    }
    int rbase = m0 + ((lane >> 4) << 3);          // lanes 0-15: M=r ; lanes 16-31: M=8+r
    f32x8 cc[4] = {c0, c1, c2, c3};
#pragma unroll
    for (int j = 0; j < 4; ++j) {
        int col = colb + j * 16;
        float bv = bias ? bias[col] : 0.0f;
#pragma unroll
        for (int r = 0; r < 8; ++r) {
            float val = cc[j][r] + bv;
            if (act == 1) val = tanhf(val);
            C[(size_t)(rbase + r) * Nd + col] = val;
        }
    }
}

// ---------------- same GEMM but bf16 output (nodes_proj; halves decode bandwidth)
__global__ __launch_bounds__(256) void
wmma_gemm_bf16out(const float* __restrict__ A, const u16* __restrict__ Bt,
                  u16* __restrict__ C, int M, int Nd) {
    int wave = threadIdx.x >> 5, lane = threadIdx.x & 31;
    int bandsN = Nd >> 6;
    int tile = blockIdx.x * 8 + wave;
    if (tile >= (M >> 4) * bandsN) return;
    int m0 = (tile / bandsN) << 4, n0 = (tile % bandsN) << 6;
    int row  = m0 + (lane & 15);
    int colb = n0 + (lane & 15);
    int koff = (lane >> 4) << 3;
    f32x8 c0 = {}, c1 = {}, c2 = {}, c3 = {};
#pragma unroll 2
    for (int k0 = 0; k0 < Kc; k0 += 32) {
        bf16x16 a  = load_a_frag_f32(A, row, k0 + koff);
        bf16x16 b0 = load_b_frag_bf16(Bt, colb,      k0 + koff);
        bf16x16 b1 = load_b_frag_bf16(Bt, colb + 16, k0 + koff);
        bf16x16 b2 = load_b_frag_bf16(Bt, colb + 32, k0 + koff);
        bf16x16 b3 = load_b_frag_bf16(Bt, colb + 48, k0 + koff);
        c0 = __builtin_amdgcn_wmma_f32_16x16x32_bf16(false, a, false, b0, (short)0, c0, false, false);
        c1 = __builtin_amdgcn_wmma_f32_16x16x32_bf16(false, a, false, b1, (short)0, c1, false, false);
        c2 = __builtin_amdgcn_wmma_f32_16x16x32_bf16(false, a, false, b2, (short)0, c2, false, false);
        c3 = __builtin_amdgcn_wmma_f32_16x16x32_bf16(false, a, false, b3, (short)0, c3, false, false);
    }
    int rbase = m0 + ((lane >> 4) << 3);
    f32x8 cc[4] = {c0, c1, c2, c3};
#pragma unroll
    for (int j = 0; j < 4; ++j) {
        int col = colb + j * 16;
#pragma unroll
        for (int r = 0; r < 8; ++r)
            C[(size_t)(rbase + r) * Nd + col] = f32_to_bf16(cc[j][r]);
    }
}

// ---------------- weight transpose + f32->bf16 (once per call)
__global__ void transpose_to_bf16(const float* __restrict__ W, u16* __restrict__ Wt, int K, int Nd) {
    int idx = blockIdx.x * blockDim.x + threadIdx.x;
    if (idx >= K * Nd) return;
    int k = idx / Nd, n = idx % Nd;
    Wt[(size_t)n * K + k] = f32_to_bf16(W[(size_t)k * Nd + n]);
}

__global__ void fillk(float* p, float v, int n) {
    int i = blockIdx.x * blockDim.x + threadIdx.x;
    if (i < n) p[i] = v;
}

// context[b,e] = mean_n nodes[b,n,e]   (64 blocks x 512 threads, coalesced)
__global__ void context_kernel(const float* __restrict__ nodes, float* __restrict__ context) {
    int b = blockIdx.x, e = threadIdx.x;
    float acc = 0.f;
    for (int n = 0; n < Nn; ++n) acc += nodes[((size_t)b * Nn + n) * En + e];
    context[b * En + e] = acc * (1.0f / (float)Nn);
}

// kp_mean[b,e] = sum_n used[b,n]*nodes[b,n,e] / max(sum_n used,1)
__global__ void kp_mean_kernel(const float* __restrict__ used, const float* __restrict__ nodes,
                               float* __restrict__ kp_mean) {
    int b = blockIdx.x, tid = threadIdx.x;
    __shared__ float sc[256];
    __shared__ float cntS;
    float c = 0.f;
    for (int n = tid; n < Nn; n += 256) c += used[b * Nn + n];
    sc[tid] = c; __syncthreads();
    for (int o = 128; o > 0; o >>= 1) { if (tid < o) sc[tid] += sc[tid + o]; __syncthreads(); }
    if (tid == 0) cntS = fmaxf(sc[0], 1.0f);
    __syncthreads();
    for (int e = tid; e < En; e += 256) {
        float acc = 0.f;
        for (int n = 0; n < Nn; ++n) {
            float u = used[b * Nn + n];               // 0/1, block-uniform -> skips loads
            if (u != 0.f) acc += nodes[((size_t)b * Nn + n) * En + e];
        }
        kp_mean[b * En + e] = acc / cntS;
    }
}

__global__ void elem_mul(const float* a, const float* b, float* c, int n) {
    int i = blockIdx.x * blockDim.x + threadIdx.x;
    if (i < n) c[i] = a[i] * b[i];
}

// GRU combine: h2 = (1-z)*n + z*h0 from precomputed gx[64,3H], gh[64,3H]
__global__ void gru_combine(const float* __restrict__ gx, const float* __restrict__ gh,
                            const float* __restrict__ h0, float* __restrict__ h2) {
    int i = blockIdx.x * blockDim.x + threadIdx.x;
    if (i >= Bn * Hn) return;
    int b = i / Hn, j = i % Hn;
    const float* gxb = gx + (size_t)b * 3 * Hn;
    const float* ghb = gh + (size_t)b * 3 * Hn;
    float r = sigmoidf_(gxb[j] + ghb[j]);
    float z = sigmoidf_(gxb[Hn + j] + ghb[Hn + j]);
    float n = tanhf(gxb[2 * Hn + j] + r * ghb[2 * Hn + j]);
    h2[i] = (1.0f - z) * n + z * h0[i];
}

// s[b,n] = sum_h v[h]*tanh(np_bf16[b,n,h] + q[b,h] + cov[b,n]*ac[h] + b_att[h])
// one wave per row; 8 rows/block. u16x2 loads: h = 2*lane + 64*i gives 128B/wave
// per load instruction and stride-2 (conflict-free on 64 banks) LDS table reads.
__global__ __launch_bounds__(256) void
attn_scores(const u16* __restrict__ np, const float* __restrict__ q,
            const float* __restrict__ cov, const float* __restrict__ ac,
            const float* __restrict__ batt, const float* __restrict__ vv,
            float* __restrict__ s) {
    __shared__ float qs[Hn], acs[Hn], bas[Hn], vs[Hn];
    int b = blockIdx.x / (Nn / 8);
    int row0 = blockIdx.x * 8;
    int tid = threadIdx.x;
    for (int i = tid; i < Hn; i += 256) {
        qs[i] = q[b * Hn + i]; acs[i] = ac[i]; bas[i] = batt[i]; vs[i] = vv[i];
    }
    __syncthreads();
    int wave = tid >> 5, lane = tid & 31;
    int row = row0 + wave;
    const u16* rp = np + (size_t)row * Hn;
    if (row + 8 < Bn * Nn) __builtin_prefetch(rp + 8 * Hn, 0, 1);  // global_prefetch_b8
    float cv = cov[row];
    float acc = 0.f;
#pragma unroll
    for (int i = 0; i < 8; ++i) {
        int h = (i << 6) + (lane << 1);
        u16x2 w = *(const u16x2*)(rp + h);
        float x0 = bf16_to_f32(w[0]) + qs[h]     + cv * acs[h]     + bas[h];
        float x1 = bf16_to_f32(w[1]) + qs[h + 1] + cv * acs[h + 1] + bas[h + 1];
        acc += vs[h] * tanhf(x0) + vs[h + 1] * tanhf(x1);
    }
#pragma unroll
    for (int o = 16; o > 0; o >>= 1) acc += __shfl_xor(acc, o, 32);
    if (lane == 0) s[row] = acc;
}

// per-batch: log-softmax stats, argmax (first-max tie break), NLL, state updates, kps write
__global__ void decode_finish(const float* __restrict__ s, const float* __restrict__ nodes,
                              const int* __restrict__ labels, float* __restrict__ used,
                              float* __restrict__ cov, float* __restrict__ inp,
                              float* __restrict__ nllv, float* __restrict__ out,
                              int l, int t) {
    int b = blockIdx.x, tid = threadIdx.x;
    __shared__ float sm[256]; __shared__ int si[256]; __shared__ float ss[256];
    const float* sb = s + (size_t)b * Nn;
    float m = -3.4e38f; int mi = Nn;
    for (int n = tid; n < Nn; n += 256) {
        float x = sb[n];
        if (x > m || (x == m && n < mi)) { m = x; mi = n; }
    }
    sm[tid] = m; si[tid] = mi; __syncthreads();
    for (int o = 128; o > 0; o >>= 1) {
        if (tid < o) {
            float a = sm[tid], c = sm[tid + o]; int ai = si[tid], ci = si[tid + o];
            if (c > a || (c == a && ci < ai)) { sm[tid] = c; si[tid] = ci; }
        }
        __syncthreads();
    }
    float mx = sm[0]; int wid = si[0];
    float se = 0.f;
    for (int n = tid; n < Nn; n += 256) se += __expf(sb[n] - mx);
    ss[tid] = se; __syncthreads();
    for (int o = 128; o > 0; o >>= 1) { if (tid < o) ss[tid] += ss[tid + o]; __syncthreads(); }
    if (tid == 0) {
        float logZ = mx + __logf(ss[0]);
        int tgt = labels[((size_t)b * KP + l) * Tn + t];
        float nll = 0.f, val = 0.f;
        if (tgt >= 0) {
            int tc = tgt > Nn - 1 ? Nn - 1 : tgt;
            nll = -(sb[tc] - logZ); val = 1.f;
        }
        nllv[b * 2] = nll; nllv[b * 2 + 1] = val;
        out[(size_t)(l * Bn + b) * (Tn + 1) + t] = (float)wid;
        if (t == Tn - 1) out[(size_t)(l * Bn + b) * (Tn + 1) + Tn] = (float)wid;
        used[(size_t)b * Nn + wid] = 1.0f;
        if (t < Tn - 1) cov[(size_t)b * Nn + wid] += 1.0f;
    }
    __syncthreads();
    if (t < Tn - 1) {
        const float* src = nodes + ((size_t)b * Nn + wid) * En;
        for (int e = tid; e < En; e += 256) inp[b * En + e] = src[e];
    }
}

// loss += sum_b(nll)/max(sum_b(valid),1)
__global__ void loss_accum(const float* __restrict__ nllv, float* __restrict__ loss) {
    __shared__ float sn[64], sv[64];
    int tid = threadIdx.x;
    sn[tid] = nllv[tid * 2]; sv[tid] = nllv[tid * 2 + 1];
    __syncthreads();
    for (int o = 32; o > 0; o >>= 1) { if (tid < o) { sn[tid] += sn[tid + o]; sv[tid] += sv[tid + o]; } __syncthreads(); }
    if (tid == 0) loss[0] += sn[0] / fmaxf(sv[0], 1.0f);
}

// ---------------------------------------------------------------------------
extern "C" void kernel_launch(void* const* d_in, const int* in_sizes, int n_in,
                              void* d_out, int out_size, void* d_ws, size_t ws_size,
                              hipStream_t stream) {
    (void)in_sizes; (void)n_in; (void)out_size; (void)ws_size;
    const float* nodes = (const float*)d_in[0];
    const int*   labels= (const int*)  d_in[1];
    const float* yW_w  = (const float*)d_in[2];
    const float* yW_b  = (const float*)d_in[3];
    const float* hW_w  = (const float*)d_in[4];
    const float* hW_b  = (const float*)d_in[5];
    const float* W_ih  = (const float*)d_in[6];
    const float* W_hh  = (const float*)d_in[7];
    const float* b_ih  = (const float*)d_in[8];
    const float* b_hh  = (const float*)d_in[9];
    const float* An    = (const float*)d_in[10];
    const float* Ah    = (const float*)d_in[11];
    const float* ac    = (const float*)d_in[12];
    const float* v     = (const float*)d_in[13];
    const float* b_att = (const float*)d_in[14];
    float* out = (float*)d_out;

    char* ws = (char*)d_ws;
    size_t cur = 0;
    auto alloc = [&](size_t bytes) -> void* {
        void* p = ws + cur;
        cur = (cur + bytes + 255) & ~(size_t)255;
        return p;
    };
    u16*   npB    = (u16*)  alloc((size_t)Bn * Nn * Hn * sizeof(u16));   // 128 MB
    u16*   AnT    = (u16*)  alloc((size_t)Hn * En * sizeof(u16));
    u16*   AhT    = (u16*)  alloc((size_t)Hn * Hn * sizeof(u16));
    u16*   yWt    = (u16*)  alloc((size_t)En * En * sizeof(u16));
    u16*   hWt    = (u16*)  alloc((size_t)Hn * En * sizeof(u16));
    u16*   WihT   = (u16*)  alloc((size_t)3 * Hn * En * sizeof(u16));
    u16*   WhhT   = (u16*)  alloc((size_t)3 * Hn * Hn * sizeof(u16));
    float* contextW=(float*)alloc((size_t)Bn * En * sizeof(float));
    float* inp    = (float*)alloc((size_t)Bn * En * sizeof(float));
    float* h0     = (float*)alloc((size_t)Bn * Hn * sizeof(float));
    float* h2     = (float*)alloc((size_t)Bn * Hn * sizeof(float));
    float* gx     = (float*)alloc((size_t)Bn * 3 * Hn * sizeof(float));
    float* gh     = (float*)alloc((size_t)Bn * 3 * Hn * sizeof(float));
    float* q      = (float*)alloc((size_t)Bn * Hn * sizeof(float));
    float* cy     = (float*)alloc((size_t)Bn * En * sizeof(float));
    float* kp_mean= (float*)alloc((size_t)Bn * En * sizeof(float));
    float* sBuf   = (float*)alloc((size_t)Bn * Nn * sizeof(float));
    float* cov    = (float*)alloc((size_t)Bn * Nn * sizeof(float));
    float* used   = (float*)alloc((size_t)Bn * Nn * sizeof(float));
    float* nllv   = (float*)alloc((size_t)Bn * 2 * sizeof(float));

    auto cdiv = [](int a, int b) { return (a + b - 1) / b; };

    // one-time conversions
    transpose_to_bf16<<<cdiv(En * Hn, 256), 256, 0, stream>>>(An, AnT, En, Hn);
    transpose_to_bf16<<<cdiv(Hn * Hn, 256), 256, 0, stream>>>(Ah, AhT, Hn, Hn);
    transpose_to_bf16<<<cdiv(En * En, 256), 256, 0, stream>>>(yW_w, yWt, En, En);
    transpose_to_bf16<<<cdiv(En * Hn, 256), 256, 0, stream>>>(hW_w, hWt, En, Hn);
    transpose_to_bf16<<<cdiv(En * 3 * Hn, 256), 256, 0, stream>>>(W_ih, WihT, En, 3 * Hn);
    transpose_to_bf16<<<cdiv(Hn * 3 * Hn, 256), 256, 0, stream>>>(W_hh, WhhT, Hn, 3 * Hn);

    context_kernel<<<Bn, En, 0, stream>>>(nodes, contextW);

    // nodes_proj = nodes @ An  (131072x512x512, bf16 out)
    {
        int tiles = ((Bn * Nn) >> 4) * (Hn >> 6);
        wmma_gemm_bf16out<<<cdiv(tiles, 8), 256, 0, stream>>>(nodes, AnT, npB, Bn * Nn, Hn);
    }

    fillk<<<cdiv(Bn * Nn, 256), 256, 0, stream>>>(used, 0.f, Bn * Nn);
    fillk<<<cdiv(Bn * Nn, 256), 256, 0, stream>>>(cov, 0.f, Bn * Nn);
    fillk<<<1, 1, 0, stream>>>(out + KP * Bn * (Tn + 1), 0.f, 1);   // loss = 0

    auto gemm = [&](const float* A, const u16* Bt, const float* bias, float* C,
                    int M, int Nd, int act) {
        int tiles = (M >> 4) * (Nd >> 6);
        wmma_gemm_f32<<<cdiv(tiles, 8), 256, 0, stream>>>(A, Bt, bias, C, M, Nd, act);
    };

    for (int l = 0; l < KP; ++l) {
        if (l == 0) {
            fillk<<<cdiv(Bn * En, 256), 256, 0, stream>>>(inp, 0.f, Bn * En);
            gemm(contextW, hWt, hW_b, h0, Bn, Hn, 1);                 // h0 = tanh(ctx@hW + b)
        } else {
            kp_mean_kernel<<<Bn, 256, 0, stream>>>(used, nodes, kp_mean);
            elem_mul<<<cdiv(Bn * En, 256), 256, 0, stream>>>(contextW, kp_mean, cy, Bn * En);
            gemm(cy, yWt, yW_b, inp, Bn, En, 0);                       // inp = cy@yW + b
            gemm(cy, hWt, hW_b, h0, Bn, Hn, 1);                        // h0 = tanh(cy@hW + b)
        }
        gemm(h0, WhhT, b_hh, gh, Bn, 3 * Hn, 0);                       // hoisted: h0 const in l

        for (int t = 0; t < Tn; ++t) {
            gemm(inp, WihT, b_ih, gx, Bn, 3 * Hn, 0);
            gru_combine<<<cdiv(Bn * Hn, 256), 256, 0, stream>>>(gx, gh, h0, h2);
            gemm(h2, AhT, nullptr, q, Bn, Hn, 0);                      // q = h2@Ah
            attn_scores<<<(Bn * Nn) / 8, 256, 0, stream>>>(npB, q, cov, ac, b_att, v, sBuf);
            decode_finish<<<Bn, 256, 0, stream>>>(sBuf, nodes, labels, used, cov, inp,
                                                  nllv, out, l, t);
            loss_accum<<<1, 64, 0, stream>>>(nllv, out + KP * Bn * (Tn + 1));
        }
    }
}
